// LLaMAAttention_59279138619557
// MI455X (gfx1250) — compile-verified, matrix operand order checked by reading, not run on hardware
//
#include <hip/hip_runtime.h>

// ---------------------------------------------------------------------------
// LLaMA attention block on gfx1250 (MI455X).
// All GEMMs + attention matmuls via v_wmma_f32_16x16x32_bf16 (fp32 accum).
// Workspace requirement: ~192 MB (x_bf16 16MB, 4 weights 128MB, Q/K/V 48MB;
// attention output aliases x_bf16).
// ---------------------------------------------------------------------------

#define S_LEN 2048
#define HID   4096
#define NH    32
#define HD    128

typedef __attribute__((ext_vector_type(16))) __bf16 bf16x16;
typedef __attribute__((ext_vector_type(8)))  float  f32x8;

union Frag { bf16x16 v; uint4 q[2]; };
union U4   { uint4 q; unsigned short s[8]; };

__device__ __forceinline__ unsigned short f2bf(float f) {
  union { float f; unsigned u; } x; x.f = f;
  unsigned r = x.u + 0x7FFFu + ((x.u >> 16) & 1u);   // round-to-nearest-even
  return (unsigned short)(r >> 16);
}
__device__ __forceinline__ float bf2f(unsigned short h) {
  union { unsigned u; float f; } x; x.u = ((unsigned)h) << 16;
  return x.f;
}

// ---------------------------------------------------------------------------
// fp32 -> bf16 bit conversion
// ---------------------------------------------------------------------------
__global__ __launch_bounds__(256)
void cvt_f32_to_bf16(const float* __restrict__ in, unsigned short* __restrict__ out, int n)
{
  int i = (blockIdx.x * 256 + threadIdx.x) * 4;
  if (i + 3 < n) {
    float4 v = *(const float4*)(in + i);
    out[i + 0] = f2bf(v.x);
    out[i + 1] = f2bf(v.y);
    out[i + 2] = f2bf(v.z);
    out[i + 3] = f2bf(v.w);
  }
}

// ---------------------------------------------------------------------------
// C[M,N] = A[M,K] * B[N,K]^T   (bf16 in, fp32 accum, bf16 or fp32 out)
// Block tile 128x128, 8 waves (2x4), wave tile 64x32 -> 4x2 WMMA accumulators.
// ---------------------------------------------------------------------------
__global__ __launch_bounds__(256)
void gemm_nt_bf16(const unsigned short* __restrict__ A,
                  const unsigned short* __restrict__ Bm,
                  float* __restrict__ Cf, unsigned short* __restrict__ Cb,
                  int M, int N, int K)
{
  __shared__ __align__(16) unsigned short As[128 * 32];
  __shared__ __align__(16) unsigned short Bs[128 * 32];

  const int tid  = threadIdx.x;
  const int wave = tid >> 5, lane = tid & 31;
  const int half = lane >> 4, ln = lane & 15;
  const int k0   = half * 8;                 // A/B fragment K-offset pattern
  const int mbase = blockIdx.y * 128;
  const int nbase = blockIdx.x * 128;
  const int wm = wave >> 2, wn = wave & 3;   // 2x4 wave grid

  f32x8 acc[4][2];
#pragma unroll
  for (int i = 0; i < 4; i++)
#pragma unroll
    for (int j = 0; j < 2; j++) acc[i][j] = {};

  for (int kk = 0; kk < K; kk += 32) {
    // cooperative LDS fill: 512 x uint4 per 128x32 tile
    const uint4* Ag = (const uint4*)A;
    const uint4* Bg = (const uint4*)Bm;
    uint4* AsU = (uint4*)As;
    uint4* BsU = (uint4*)Bs;
#pragma unroll
    for (int t = 0; t < 2; t++) {
      int i   = tid + t * 256;
      int row = i >> 2;                      // 4 uint4 (32 elems) per row
      int c8  = i & 3;
      AsU[i] = Ag[((size_t)(mbase + row) * K + kk) / 8 + c8];
      BsU[i] = Bg[((size_t)(nbase + row) * K + kk) / 8 + c8];
    }
    __syncthreads();

    if (kk + 32 < K) {                       // prefetch next K-slab
      __builtin_prefetch(&A[(size_t)(mbase + (tid >> 1)) * K + kk + 32 + (tid & 1) * 16], 0, 1);
      __builtin_prefetch(&Bm[(size_t)(nbase + (tid >> 1)) * K + kk + 32 + (tid & 1) * 16], 0, 1);
    }

    Frag af[4], bfr[2];
#pragma unroll
    for (int mi = 0; mi < 4; mi++) {
      const unsigned short* p = &As[(wm * 64 + mi * 16 + ln) * 32 + k0];
      af[mi].q[0] = *(const uint4*)p;
      af[mi].q[1] = *(const uint4*)(p + 16);
    }
#pragma unroll
    for (int ni = 0; ni < 2; ni++) {
      const unsigned short* p = &Bs[(wn * 32 + ni * 16 + ln) * 32 + k0];
      bfr[ni].q[0] = *(const uint4*)p;
      bfr[ni].q[1] = *(const uint4*)(p + 16);
    }
#pragma unroll
    for (int mi = 0; mi < 4; mi++)
#pragma unroll
      for (int ni = 0; ni < 2; ni++)
        acc[mi][ni] = __builtin_amdgcn_wmma_f32_16x16x32_bf16(
            false, af[mi].v, false, bfr[ni].v, (short)0, acc[mi][ni], false, false);
    __syncthreads();
  }

  // epilogue: C layout = VGPR r -> rows r / r+8 per half, N = lane%16
#pragma unroll
  for (int mi = 0; mi < 4; mi++)
#pragma unroll
    for (int ni = 0; ni < 2; ni++)
#pragma unroll
      for (int r = 0; r < 8; r++) {
        int row = mbase + wm * 64 + mi * 16 + r + half * 8;
        int col = nbase + wn * 32 + ni * 16 + ln;
        float v = acc[mi][ni][r];
        if (Cb) Cb[(size_t)row * N + col] = f2bf(v);
        else    Cf[(size_t)row * N + col] = v;
      }
}

// ---------------------------------------------------------------------------
// RoPE in place on bf16 Q and K: [S, H, D] pairs, angles indexed [h, d/2]
// ---------------------------------------------------------------------------
__global__ __launch_bounds__(256)
void rope_kernel(unsigned short* __restrict__ Q, unsigned short* __restrict__ Kc,
                 const float* __restrict__ fcos, const float* __restrict__ fsin)
{
  int gid = blockIdx.x * 256 + threadIdx.x;      // over S*H*(D/2) = 2048*32*64
  if (gid >= S_LEN * NH * (HD / 2)) return;
  int i = gid & 63;
  int h = (gid >> 6) & 31;
  int s = gid >> 11;
  float c = fcos[h * 64 + i], sn = fsin[h * 64 + i];
  size_t off = (size_t)s * HID + h * HD + 2 * i;
  {
    float re = bf2f(Q[off]), im = bf2f(Q[off + 1]);
    Q[off]     = f2bf(re * c - im * sn);
    Q[off + 1] = f2bf(re * sn + im * c);
  }
  {
    float re = bf2f(Kc[off]), im = bf2f(Kc[off + 1]);
    Kc[off]     = f2bf(re * c - im * sn);
    Kc[off + 1] = f2bf(re * sn + im * c);
  }
}

// ---------------------------------------------------------------------------
// Flash attention, causal. Grid = (S/128 q-tiles, H). 8 waves/block, each wave
// owns 16 query rows. K tile [32x128] + V tile transposed [128x32] in LDS.
// ---------------------------------------------------------------------------
__global__ __launch_bounds__(256)
void attn_kernel(const unsigned short* __restrict__ Qg,
                 const unsigned short* __restrict__ Kg,
                 const unsigned short* __restrict__ Vg,
                 unsigned short* __restrict__ Og)
{
  __shared__ __align__(16) unsigned short Ks[32 * 128];     // [key][d]
  __shared__ __align__(16) unsigned short Vt[128 * 32];     // [d][key]
  __shared__ __align__(16) unsigned short Ps[8][16 * 32];   // per-wave P

  const int qt = blockIdx.x, h = blockIdx.y;
  const int tid = threadIdx.x, wave = tid >> 5, lane = tid & 31;
  const int half = lane >> 4, ln = lane & 15, k0 = half * 8;
  const int q0 = qt * 128 + wave * 16;
  const float scale = 0.08838834764831845f;                 // 1/sqrt(128)

  // Q A-fragments: 4 chunks of 32 along D
  Frag aq[4];
  {
    size_t base = (size_t)(q0 + ln) * HID + h * HD;
#pragma unroll
    for (int c = 0; c < 4; c++) {
      const unsigned short* p = Qg + base + c * 32 + k0;
      aq[c].q[0] = *(const uint4*)p;
      aq[c].q[1] = *(const uint4*)(p + 16);
    }
  }

  f32x8 o[8];
#pragma unroll
  for (int t = 0; t < 8; t++) o[t] = {};
  float rowmax[8], rowsum[8];
#pragma unroll
  for (int r = 0; r < 8; r++) { rowmax[r] = -1e30f; rowsum[r] = 0.f; }

  const int nkb = (qt + 1) * 4;                 // causal: keys < (qt+1)*128
  for (int kb = 0; kb < nkb; kb++) {
    const int kbase = kb * 32;
    // ---- cooperative K / V(T) tile load ----
    {
      uint4* KsU = (uint4*)Ks;
#pragma unroll
      for (int t = 0; t < 2; t++) {
        int i = tid + t * 256;                  // 512 chunks of 8 elems
        int kk = i >> 4, d8 = i & 15;
        KsU[i] = *(const uint4*)(Kg + (size_t)(kbase + kk) * HID + h * HD + d8 * 8);
      }
#pragma unroll
      for (int t = 0; t < 2; t++) {
        int i = tid + t * 256;
        int kk = i >> 4, d0 = (i & 15) * 8;
        U4 u;
        u.q = *(const uint4*)(Vg + (size_t)(kbase + kk) * HID + h * HD + d0);
#pragma unroll
        for (int j = 0; j < 8; j++) Vt[(d0 + j) * 32 + kk] = u.s[j];
      }
    }
    __syncthreads();

    if (kbase <= q0 + 15) {                     // wave-uniform: EXEC stays full
      // ---- S = Q K^T (16x32 scores) ----
      f32x8 sAcc[2];
#pragma unroll
      for (int nt = 0; nt < 2; nt++) {
        sAcc[nt] = {};
#pragma unroll
        for (int c = 0; c < 4; c++) {
          Frag bk;
          const unsigned short* p = &Ks[(nt * 16 + ln) * 128 + c * 32 + k0];
          bk.q[0] = *(const uint4*)p;
          bk.q[1] = *(const uint4*)(p + 16);
          sAcc[nt] = __builtin_amdgcn_wmma_f32_16x16x32_bf16(
              false, aq[c].v, false, bk.v, (short)0, sAcc[nt], false, false);
        }
      }
      // ---- scale + causal mask + online softmax ----
      float sc[2][8];
#pragma unroll
      for (int nt = 0; nt < 2; nt++)
#pragma unroll
        for (int r = 0; r < 8; r++) {
          int kidx = kbase + nt * 16 + ln;
          int qidx = q0 + r + half * 8;
          float v = sAcc[nt][r] * scale;
          if (kidx > qidx) v += -1e9f;
          sc[nt][r] = v;
        }
#pragma unroll
      for (int r = 0; r < 8; r++) {
        float t = fmaxf(sc[0][r], sc[1][r]);
#pragma unroll
        for (int m = 1; m < 16; m <<= 1) t = fmaxf(t, __shfl_xor(t, m, 32));
        float mn = fmaxf(rowmax[r], t);
        float alpha = __expf(rowmax[r] - mn);
        rowmax[r] = mn;
        float p0 = __expf(sc[0][r] - mn);
        float p1 = __expf(sc[1][r] - mn);
        sc[0][r] = p0; sc[1][r] = p1;
        float ps = p0 + p1;
#pragma unroll
        for (int m = 1; m < 16; m <<= 1) ps += __shfl_xor(ps, m, 32);
        rowsum[r] = rowsum[r] * alpha + ps;
#pragma unroll
        for (int t2 = 0; t2 < 8; t2++) o[t2][r] *= alpha;
      }
      // ---- P -> LDS -> A-fragment relayout ----
#pragma unroll
      for (int nt = 0; nt < 2; nt++)
#pragma unroll
        for (int r = 0; r < 8; r++)
          Ps[wave][(r + half * 8) * 32 + nt * 16 + ln] = f2bf(sc[nt][r]);
      asm volatile("s_wait_dscnt 0" ::: "memory");   // intra-wave LDS RAW
      Frag pa;
      {
        const unsigned short* p = &Ps[wave][ln * 32 + k0];
        pa.q[0] = *(const uint4*)p;
        pa.q[1] = *(const uint4*)(p + 16);
      }
      // ---- O += P V ----
#pragma unroll
      for (int dt = 0; dt < 8; dt++) {
        Frag bv;
        const unsigned short* p = &Vt[(dt * 16 + ln) * 32 + k0];
        bv.q[0] = *(const uint4*)p;
        bv.q[1] = *(const uint4*)(p + 16);
        o[dt] = __builtin_amdgcn_wmma_f32_16x16x32_bf16(
            false, pa.v, false, bv.v, (short)0, o[dt], false, false);
      }
    }
    __syncthreads();
  }

  // ---- epilogue: O / rowsum -> bf16 [S, H*D] ----
  float rinv[8];
#pragma unroll
  for (int r = 0; r < 8; r++) rinv[r] = 1.f / rowsum[r];
#pragma unroll
  for (int dt = 0; dt < 8; dt++)
#pragma unroll
    for (int r = 0; r < 8; r++) {
      int row = q0 + r + half * 8;
      int col = h * HD + dt * 16 + ln;
      Og[(size_t)row * HID + col] = f2bf(o[dt][r] * rinv[r]);
    }
}

// ---------------------------------------------------------------------------
// Host-side orchestration
// ---------------------------------------------------------------------------
extern "C" void kernel_launch(void* const* d_in, const int* in_sizes, int n_in,
                              void* d_out, int out_size, void* d_ws, size_t ws_size,
                              hipStream_t stream)
{
  const float* x  = (const float*)d_in[0];
  const float* fc = (const float*)d_in[1];
  const float* fs = (const float*)d_in[2];
  // d_in[3] = mask, handled analytically (causal)
  const float* wq = (const float*)d_in[4];
  const float* wk = (const float*)d_in[5];
  const float* wv = (const float*)d_in[6];
  const float* wo = (const float*)d_in[7];
  float* out = (float*)d_out;

  const size_t SH = (size_t)S_LEN * HID;   // 8,388,608
  const size_t WW = (size_t)HID * HID;     // 16,777,216

  char* p = (char*)d_ws;
  auto take = [&](size_t bytes) { char* r = p; p += (bytes + 255) & ~(size_t)255; return r; };
  unsigned short* xb  = (unsigned short*)take(SH * 2);
  unsigned short* wqb = (unsigned short*)take(WW * 2);
  unsigned short* wkb = (unsigned short*)take(WW * 2);
  unsigned short* wvb = (unsigned short*)take(WW * 2);
  unsigned short* wob = (unsigned short*)take(WW * 2);
  unsigned short* qb  = (unsigned short*)take(SH * 2);
  unsigned short* kb  = (unsigned short*)take(SH * 2);
  unsigned short* vb  = (unsigned short*)take(SH * 2);
  unsigned short* ab  = xb;   // x's bf16 copy is dead after QKV -> reuse

  cvt_f32_to_bf16<<<(int)(SH / 4 / 256), 256, 0, stream>>>(x,  xb,  (int)SH);
  cvt_f32_to_bf16<<<(int)(WW / 4 / 256), 256, 0, stream>>>(wq, wqb, (int)WW);
  cvt_f32_to_bf16<<<(int)(WW / 4 / 256), 256, 0, stream>>>(wk, wkb, (int)WW);
  cvt_f32_to_bf16<<<(int)(WW / 4 / 256), 256, 0, stream>>>(wv, wvb, (int)WW);
  cvt_f32_to_bf16<<<(int)(WW / 4 / 256), 256, 0, stream>>>(wo, wob, (int)WW);

  dim3 gg(HID / 128, S_LEN / 128);   // (32, 16)
  gemm_nt_bf16<<<gg, 256, 0, stream>>>(xb, wqb, nullptr, qb, S_LEN, HID, HID);
  gemm_nt_bf16<<<gg, 256, 0, stream>>>(xb, wkb, nullptr, kb, S_LEN, HID, HID);
  gemm_nt_bf16<<<gg, 256, 0, stream>>>(xb, wvb, nullptr, vb, S_LEN, HID, HID);

  rope_kernel<<<(S_LEN * NH * (HD / 2)) / 256, 256, 0, stream>>>(qb, kb, fc, fs);

  attn_kernel<<<dim3(S_LEN / 128, NH), 256, 0, stream>>>(qb, kb, vb, ab);

  gemm_nt_bf16<<<gg, 256, 0, stream>>>(ab, wob, out, nullptr, S_LEN, HID, HID);
}
